// HeteroNeighborhoodAttention_24232205484251
// MI455X (gfx1250) — compile-verified, matrix operand order checked by reading, not run on hardware
//
#include <hip/hip_runtime.h>
#include <hip/hip_bf16.h>
#include <stdint.h>

// ---------------------------------------------------------------------------
// HeteroNeighborhoodAttention for MI455X (gfx1250, wave32, WMMA bf16)
// Round 3: native bf16 converts (v_cvt_bf16_f32 / v_cvt_pk_bf16_f32) instead
// of software RNE rounding; otherwise same structure as round 2.
// ---------------------------------------------------------------------------

typedef __attribute__((ext_vector_type(16))) __bf16          v16bf;
typedef __attribute__((ext_vector_type(8)))  float           v8f;
typedef __attribute__((ext_vector_type(8)))  unsigned short  u16x8;

#define OUT_DIM 128
#define N_HEADS 8
#define SRC_DIM 64
#define DST_DIM 64
#define EDGE_DIM 16
#define IN_DIM 144
#define IN_PAD 160          // 144 padded to 5 * 32
#define EK_WAVES 4          // waves per block; 16 edges/rows per wave

union Frag16 { v16bf v; u16x8 h2[2]; unsigned short u[16]; };

// native hardware converts (backend selects v_cvt_bf16_f32 / v_cvt_pk_bf16_f32)
__device__ __forceinline__ unsigned short f2bf(float f) {
  union { __bf16 b; unsigned short u; } cv;
  cv.b = (__bf16)f;
  return cv.u;
}
__device__ __forceinline__ float bf2f(unsigned short h) {
  union { unsigned short u; __bf16 b; } cv;
  cv.u = h;
  return (float)cv.b;
}
// order-preserving float <-> uint encoding for atomicMax-based segment max
__device__ __forceinline__ unsigned int fenc(float f) {
  unsigned int u = __float_as_uint(f);
  return (u & 0x80000000u) ? ~u : (u | 0x80000000u);
}
__device__ __forceinline__ float fdec(unsigned int u) {
  return __uint_as_float((u & 0x80000000u) ? (u & 0x7FFFFFFFu) : ~u);
}
#define ENC_NEG_INF 0x007FFFFFu   // fenc(-inf)

// ---------------------------------------------------------------------------
// One-wave residual stack on a 16-row tile:
//   h   = relu(x @ w0 + b0)     x: [16, KROW] bf16 in LDS (zero padded)
//   res = h + h @ w1 + b1
// Weights are pre-swizzled fragment-linear: for tile (kt, n), lane l holds 16
// contiguous bf16 at ((kt*8 + n)*32 + l)*16 -> one 32B vector load per B-frag.
// res returned in WMMA C-layout: res[n][j] is element (M = j + 8*(lane>=16),
// N = n*16 + (lane&15)).
// ---------------------------------------------------------------------------
template<int KROW, int KT0>
__device__ __forceinline__ void res_block(
    const unsigned short* __restrict__ xs,   // [16][KROW] this wave's tile
    unsigned short* __restrict__ hs,         // [16][128] hidden scratch
    const unsigned short* __restrict__ w0sw, const float* __restrict__ bias0,
    const unsigned short* __restrict__ w1sw, const float* __restrict__ bias1,
    int lane, float res[8][8])
{
  const int r    = lane & 15;         // A row within half / B,C column
  const int kh   = (lane >> 4) << 3;  // K sub-half select: 0 or 8
  const int mrow = (lane >> 4) << 3;  // C-layout M base: 0 or 8
  Frag16 a, b;
  const v8f vzero = {0.f,0.f,0.f,0.f,0.f,0.f,0.f,0.f};
  v8f acc[8];
#pragma unroll
  for (int n = 0; n < 8; n++) acc[n] = vzero;

  // ---- layer 0: x @ w0 ----
#pragma unroll
  for (int kt = 0; kt < KT0; kt++) {
    const int k0 = kt * 32;
    a.h2[0] = *(const u16x8*)(xs + r * KROW + k0 + kh);        // ds_load_b128
    a.h2[1] = *(const u16x8*)(xs + r * KROW + k0 + 16 + kh);   // ds_load_b128
#pragma unroll
    for (int n = 0; n < 8; n++) {
      b.v = *(const v16bf*)(w0sw + (((kt * 8 + n) * 32 + lane) << 4));
      acc[n] = __builtin_amdgcn_wmma_f32_16x16x32_bf16(
          false, a.v, false, b.v, (short)0, acc[n], false, false);
    }
  }

  // ---- bias + relu, stage hidden tile to LDS (bf16) ----
#pragma unroll
  for (int n = 0; n < 8; n++) {
    const float bb = bias0[n * 16 + r];
#pragma unroll
    for (int j = 0; j < 8; j++) {
      float h = acc[n][j] + bb;
      h = h > 0.f ? h : 0.f;
      acc[n][j] = h;                                   // keep fp32 residual
      hs[(mrow + j) * OUT_DIM + n * 16 + r] = f2bf(h);
    }
  }
  __syncthreads();

  // ---- layer 1: h @ w1 ----
  v8f acc2[8];
#pragma unroll
  for (int n = 0; n < 8; n++) acc2[n] = vzero;
#pragma unroll
  for (int kt = 0; kt < 4; kt++) {
    const int k0 = kt * 32;
    a.h2[0] = *(const u16x8*)(hs + r * OUT_DIM + k0 + kh);
    a.h2[1] = *(const u16x8*)(hs + r * OUT_DIM + k0 + 16 + kh);
#pragma unroll
    for (int n = 0; n < 8; n++) {
      b.v = *(const v16bf*)(w1sw + (((kt * 8 + n) * 32 + lane) << 4));
      acc2[n] = __builtin_amdgcn_wmma_f32_16x16x32_bf16(
          false, a.v, false, b.v, (short)0, acc2[n], false, false);
    }
  }

  // ---- residual + bias1 ----
#pragma unroll
  for (int n = 0; n < 8; n++) {
    const float bb = bias1[n * 16 + r];
#pragma unroll
    for (int j = 0; j < 8; j++)
      res[n][j] = acc[n][j] + acc2[n][j] + bb;
  }
  __syncthreads();   // hs may be reused by the caller
}

// ---------------------------------------------------------------------------
// Kernel: per-edge gather + K/V residual stacks + per-head scores
// v_out is stored TRANSPOSED: v_out[col * E + e] (bf16)
// ---------------------------------------------------------------------------
__global__ __launch_bounds__(EK_WAVES * 32)
void edge_kv_kernel(
    const float* __restrict__ x_src, const float* __restrict__ x_dst,
    const float* __restrict__ edge_attr,
    const int* __restrict__ src_idx, const int* __restrict__ dst_idx,
    const float* __restrict__ q,
    const unsigned short* __restrict__ kw0, const float* __restrict__ kb0,
    const unsigned short* __restrict__ kw1, const float* __restrict__ kb1,
    const unsigned short* __restrict__ vw0, const float* __restrict__ vb0,
    const unsigned short* __restrict__ vw1, const float* __restrict__ vb1,
    float* __restrict__ scores, unsigned short* __restrict__ v_out, int E)
{
  __shared__ unsigned short xs_all[EK_WAVES][16][IN_PAD];
  __shared__ unsigned short hs_all[EK_WAVES][16][OUT_DIM];
  const int wid  = threadIdx.x >> 5;
  const int lane = threadIdx.x & 31;
  const int e0   = (blockIdx.x * EK_WAVES + wid) * 16;
  unsigned short* xs = &xs_all[wid][0][0];
  unsigned short* hs = &hs_all[wid][0][0];

  __builtin_prefetch(kw0, 0, 1);   // global_prefetch_b8: weights -> caches
  __builtin_prefetch(vw0, 0, 1);
  __builtin_prefetch(kw1, 0, 1);
  __builtin_prefetch(vw1, 0, 1);

  // gather 16 edges x 160 features into LDS (bf16, zero padded);
  // edge loop outside so index loads are wave-uniform and feature loads
  // are 32-consecutive-float coalesced.
#pragma unroll 2
  for (int el = 0; el < 16; ++el) {
    const int e = e0 + el;
    const bool valid = (e < E);
    int si = 0, di = 0;
    if (valid) { si = src_idx[e]; di = dst_idx[e]; }
    for (int c = lane; c < IN_PAD; c += 32) {
      float val = 0.f;
      if (valid) {
        if (c < SRC_DIM)                 val = x_src[(long)si * SRC_DIM + c];
        else if (c < SRC_DIM + DST_DIM)  val = x_dst[(long)di * DST_DIM + (c - SRC_DIM)];
        else if (c < IN_DIM)             val = edge_attr[(long)e * EDGE_DIM + (c - SRC_DIM - DST_DIM)];
      }
      xs[el * IN_PAD + c] = f2bf(val);
    }
  }
  __syncthreads();

  const int r     = lane & 15;
  const int mbase = (lane >> 4) << 3;

  // ---- K stack -> per-head scores (head == N-tile since HEAD_D == 16) ----
  {
    float kres[8][8];
    res_block<IN_PAD, 5>(xs, hs, kw0, kb0, kw1, kb1, lane, kres);
#pragma unroll
    for (int n = 0; n < 8; n++) {
      const float qv = q[n * 16 + r] * 0.25f;  // 1/sqrt(16)
      float s[8];
#pragma unroll
      for (int j = 0; j < 8; j++) s[j] = qv * kres[n][j];
      // reduce over the 16 lanes of each half (d = 16 dot product)
#pragma unroll
      for (int m = 8; m >= 1; m >>= 1) {
#pragma unroll
        for (int j = 0; j < 8; j++) s[j] += __shfl_xor(s[j], m, 32);
      }
#pragma unroll
      for (int j = 0; j < 8; j++) {
        if (r == j) {
          const int e = e0 + mbase + j;
          if (e < E) scores[(long)e * N_HEADS + n] = s[j];
        }
      }
    }
  }

  // ---- V stack -> store v transposed (bf16, [128][E]) ----
  {
    float vres[8][8];
    res_block<IN_PAD, 5>(xs, hs, vw0, vb0, vw1, vb1, lane, vres);
    if (((E & 7) == 0) && (e0 + 16 <= E)) {
      // 8 row-adjacent values per lane -> one 16B store per n-tile
#pragma unroll
      for (int n = 0; n < 8; n++) {
        u16x8 pk;
#pragma unroll
        for (int j = 0; j < 8; j++) pk[j] = f2bf(vres[n][j]);
        *(u16x8*)(v_out + (size_t)(n * 16 + r) * E + e0 + mbase) = pk;
      }
    } else {
#pragma unroll
      for (int n = 0; n < 8; n++) {
#pragma unroll
        for (int j = 0; j < 8; j++) {
          const int e = e0 + mbase + j;
          if (e < E) v_out[(size_t)(n * 16 + r) * E + e] = f2bf(vres[n][j]);
        }
      }
    }
  }
}

// ---------------------------------------------------------------------------
// Softmax / scatter helpers
// ---------------------------------------------------------------------------
__global__ void init_kernel(unsigned int* __restrict__ segmax,
                            float* __restrict__ denom,
                            float* __restrict__ aggr, int nseg, int naggr)
{
  const int i = blockIdx.x * blockDim.x + threadIdx.x;
  if (i < nseg) { segmax[i] = ENC_NEG_INF; denom[i] = 0.f; }
  if (i < naggr) aggr[i] = 0.f;
}

__global__ void seg_max_kernel(const float* __restrict__ scores,
                               const int* __restrict__ dst_idx,
                               unsigned int* __restrict__ segmax, int EH)
{
  const int i = blockIdx.x * blockDim.x + threadIdx.x;
  if (i >= EH) return;
  const int e = i >> 3, h = i & 7;
  atomicMax(&segmax[(long)dst_idx[e] * N_HEADS + h], fenc(scores[i]));
}

__global__ void exp_sum_kernel(float* __restrict__ scores,
                               const int* __restrict__ dst_idx,
                               const unsigned int* __restrict__ segmax,
                               float* __restrict__ denom, int EH)
{
  const int i = blockIdx.x * blockDim.x + threadIdx.x;
  if (i >= EH) return;
  const int e = i >> 3, h = i & 7;
  const int d = dst_idx[e];
  float m = fdec(segmax[(long)d * N_HEADS + h]);
  if (m < -1e37f) m = 0.f;                    // non-finite seg_max -> 0
  const float ex = __expf(scores[i] - m);
  scores[i] = ex;                             // reuse buffer for exp values
  atomicAdd(&denom[(long)d * N_HEADS + h], ex);
}

// grid: (ceil(E/256), 128); blockIdx.y = output column (head = col>>4)
__global__ void aggregate_kernel(const float* __restrict__ ex,
                                 const float* __restrict__ denom,
                                 const int* __restrict__ dst_idx,
                                 const unsigned short* __restrict__ vbfT,
                                 float* __restrict__ aggr, int E)
{
  const int e = blockIdx.x * blockDim.x + threadIdx.x;
  if (e >= E) return;
  const int c = blockIdx.y;
  const int h = c >> 4;
  const int d = dst_idx[e];
  const float attn = ex[(size_t)e * N_HEADS + h] /
                     (denom[(size_t)d * N_HEADS + h] + 1e-16f);
  atomicAdd(&aggr[(size_t)d * OUT_DIM + c], attn * bf2f(vbfT[(size_t)c * E + e]));
}

// ---------------------------------------------------------------------------
// Kernel: output residual stack  out = relu(resblock(relu(aggr)))
// ---------------------------------------------------------------------------
__global__ __launch_bounds__(EK_WAVES * 32)
void out_mlp_kernel(const float* __restrict__ aggr,
                    const unsigned short* __restrict__ ow0, const float* __restrict__ ob0,
                    const unsigned short* __restrict__ ow1, const float* __restrict__ ob1,
                    float* __restrict__ out, int N)
{
  __shared__ unsigned short xs_all[EK_WAVES][16][OUT_DIM];
  __shared__ unsigned short hs_all[EK_WAVES][16][OUT_DIM];
  const int wid  = threadIdx.x >> 5;
  const int lane = threadIdx.x & 31;
  const int r0   = (blockIdx.x * EK_WAVES + wid) * 16;
  unsigned short* xs = &xs_all[wid][0][0];
  unsigned short* hs = &hs_all[wid][0][0];

  __builtin_prefetch(ow0, 0, 1);
  __builtin_prefetch(ow1, 0, 1);

  for (int idx = lane; idx < 16 * OUT_DIM; idx += 32) {
    const int rl = idx >> 7, c = idx & 127;
    const int row = r0 + rl;
    float v = 0.f;
    if (row < N) { v = aggr[(long)row * OUT_DIM + c]; v = v > 0.f ? v : 0.f; }
    xs[idx] = f2bf(v);
  }
  __syncthreads();

  float res[8][8];
  res_block<OUT_DIM, 4>(xs, hs, ow0, ob0, ow1, ob1, lane, res);

  const int r = lane & 15, mbase = (lane >> 4) << 3;
#pragma unroll
  for (int n = 0; n < 8; n++) {
#pragma unroll
    for (int j = 0; j < 8; j++) {
      const int row = r0 + mbase + j;
      float v = res[n][j];
      v = v > 0.f ? v : 0.f;                  // last_activation = True
      if (row < N) out[(long)row * OUT_DIM + n * 16 + r] = v;
    }
  }
}

// ---------------------------------------------------------------------------
// Weight conversion: fp32 -> bf16, re-laid out fragment-linear.
// Swizzled index i within one matrix region:
//   elem = i & 15, lane = (i >> 4) & 31, tile = i >> 9, n = tile & 7,
//   kt = tile >> 3
//   k   = kt*32 + ((lane>>4)<<3) + (elem&7) + ((elem>>3)<<4)
//   col = n*16 + (lane&15)
// so each lane's B-fragment is 16 contiguous bf16 (32B aligned).
// ---------------------------------------------------------------------------
__global__ void convert_weights_kernel(
    const float* __restrict__ kw0, const float* __restrict__ vw0,
    const float* __restrict__ kw1, const float* __restrict__ vw1,
    const float* __restrict__ ow0, const float* __restrict__ ow1,
    unsigned short* __restrict__ wsw)
{
  const int W0 = IN_PAD * OUT_DIM;   // 20480 = 5 kt * 8 n * 32 lanes * 16
  const int W1 = OUT_DIM * OUT_DIM;  // 16384 = 4 kt * 8 n * 32 lanes * 16
  const int total = 2 * W0 + 4 * W1;
  const int i = blockIdx.x * blockDim.x + threadIdx.x;
  if (i >= total) return;

  const float* W; int rows, o;
  if (i < W0)        { W = kw0; rows = IN_DIM; o = i; }
  else if (i < 2*W0) { W = vw0; rows = IN_DIM; o = i - W0; }
  else {
    const int t = i - 2 * W0, sel = t / W1;
    o = t % W1;
    W = (sel == 0) ? kw1 : (sel == 1) ? vw1 : (sel == 2) ? ow0 : ow1;
    rows = OUT_DIM;
  }
  const int elem = o & 15, lane = (o >> 4) & 31, tile = o >> 9;
  const int n = tile & 7, kt = tile >> 3;
  const int k   = kt * 32 + ((lane >> 4) << 3) + (elem & 7) + ((elem >> 3) << 4);
  const int col = n * 16 + (lane & 15);
  wsw[i] = f2bf((k < rows) ? W[k * OUT_DIM + col] : 0.f);
}

// ---------------------------------------------------------------------------
extern "C" void kernel_launch(void* const* d_in, const int* in_sizes, int n_in,
                              void* d_out, int out_size, void* d_ws, size_t ws_size,
                              hipStream_t stream)
{
  const float* x_src     = (const float*)d_in[0];
  const float* x_dst     = (const float*)d_in[1];
  const float* edge_attr = (const float*)d_in[2];
  const int*   edge_idx  = (const int*)d_in[3];
  const float* q         = (const float*)d_in[4];
  const float* k_w0 = (const float*)d_in[5];  const float* k_b0 = (const float*)d_in[6];
  const float* k_w1 = (const float*)d_in[7];  const float* k_b1 = (const float*)d_in[8];
  const float* v_w0 = (const float*)d_in[9];  const float* v_b0 = (const float*)d_in[10];
  const float* v_w1 = (const float*)d_in[11]; const float* v_b1 = (const float*)d_in[12];
  const float* o_w0 = (const float*)d_in[13]; const float* o_b0 = (const float*)d_in[14];
  const float* o_w1 = (const float*)d_in[15]; const float* o_b1 = (const float*)d_in[16];
  (void)n_in; (void)out_size; (void)ws_size;

  const int E = in_sizes[3] / 2;
  const int N = in_sizes[1] / DST_DIM;
  const int* src_idx = edge_idx;
  const int* dst_idx = edge_idx + E;

  // workspace carve-up (256B aligned blocks)
  char* ws = (char*)d_ws;
  size_t off = 0;
  auto carve = [&](size_t bytes) -> char* {
    char* p = ws + off;
    off = (off + bytes + 255) & ~(size_t)255;
    return p;
  };
  const int W0 = IN_PAD * OUT_DIM, W1 = OUT_DIM * OUT_DIM;
  unsigned short* wbf   = (unsigned short*)carve((size_t)(2 * W0 + 4 * W1) * 2);
  unsigned short* kw0b  = wbf;
  unsigned short* vw0b  = wbf + W0;
  unsigned short* kw1b  = wbf + 2 * W0;
  unsigned short* vw1b  = kw1b + W1;
  unsigned short* ow0b  = vw1b + W1;
  unsigned short* ow1b  = ow0b + W1;
  float*          scores = (float*)carve((size_t)E * N_HEADS * 4);
  unsigned int*   segmax = (unsigned int*)carve((size_t)N * N_HEADS * 4);
  float*          denom  = (float*)carve((size_t)N * N_HEADS * 4);
  float*          aggr   = (float*)carve((size_t)N * OUT_DIM * 4);
  unsigned short* vbfT   = (unsigned short*)carve((size_t)E * OUT_DIM * 2);

  {
    const int total = 2 * W0 + 4 * W1;
    convert_weights_kernel<<<(total + 255) / 256, 256, 0, stream>>>(
        k_w0, v_w0, k_w1, v_w1, o_w0, o_w1, wbf);
  }
  {
    const int naggr = N * OUT_DIM;
    init_kernel<<<(naggr + 255) / 256, 256, 0, stream>>>(
        segmax, denom, aggr, N * N_HEADS, naggr);
  }
  {
    const int rows_per_blk = EK_WAVES * 16;
    const int blocks = (E + rows_per_blk - 1) / rows_per_blk;
    edge_kv_kernel<<<blocks, EK_WAVES * 32, 0, stream>>>(
        x_src, x_dst, edge_attr, src_idx, dst_idx, q,
        kw0b, k_b0, kw1b, k_b1, vw0b, v_b0, vw1b, v_b1,
        scores, vbfT, E);
  }
  {
    const int EH = E * N_HEADS;
    seg_max_kernel<<<(EH + 255) / 256, 256, 0, stream>>>(scores, dst_idx, segmax, EH);
    exp_sum_kernel<<<(EH + 255) / 256, 256, 0, stream>>>(scores, dst_idx, segmax, denom, EH);
  }
  {
    dim3 grid((E + 255) / 256, OUT_DIM);
    aggregate_kernel<<<grid, 256, 0, stream>>>(scores, denom, dst_idx, vbfT, aggr, E);
  }
  {
    const int rows_per_blk = EK_WAVES * 16;
    const int blocks = (N + rows_per_blk - 1) / rows_per_blk;
    out_mlp_kernel<<<blocks, EK_WAVES * 32, 0, stream>>>(
        aggr, ow0b, o_b0, ow1b, o_b1, (float*)d_out, N);
  }
}
